// HierarchicalDecoder_40209483825774
// MI455X (gfx1250) — compile-verified, matrix operand order checked by reading, not run on hardware
//
#include <hip/hip_runtime.h>
#include <hip/hip_bf16.h>

typedef unsigned short u16;
typedef __attribute__((ext_vector_type(16))) __bf16 v16bf;
typedef __attribute__((ext_vector_type(8)))  float  v8f;
typedef __attribute__((ext_vector_type(4)))  unsigned int vu4;

union Frag { v16bf v; vu4 q[2]; };

#define U_DIM 16
#define T_DIM 512
#define COND 512
#define HID 1024
#define OUTD 16

__device__ __forceinline__ u16 f2b(float f) {
    unsigned int u = __float_as_uint(f);
    unsigned int r = (u + 0x7FFFu + ((u >> 16) & 1u)) >> 16;
    return (u16)r;
}
__device__ __forceinline__ float sigf(float x) { return 1.0f / (1.0f + __expf(-x)); }
__device__ __forceinline__ float tanhf_(float x) {
    float e = __expf(-2.0f * x);
    return (1.0f - e) / (1.0f + e);
}
__device__ __forceinline__ v8f zero8() { v8f z = {0,0,0,0,0,0,0,0}; return z; }

// ---------------------------------------------------------------------------
__global__ void zero_u32(unsigned int* __restrict__ p, int n) {
    int i = blockIdx.x * 64 + threadIdx.x;
    if (i < n) p[i] = 0u;
}

__global__ void cvt_f32_bf16(const float* __restrict__ in, u16* __restrict__ out, int n) {
    int i = blockIdx.x * 256 + threadIdx.x;
    if (i < n) out[i] = f2b(in[i]);
}

// Pack z [T,512] -> A0 bf16 [U*T, 32] with row m = u*T + t, k contiguous.
__global__ void pack_z(const float* __restrict__ z, u16* __restrict__ a0) {
    int i = blockIdx.x * 256 + threadIdx.x;   // over U*T*32
    if (i >= U_DIM * T_DIM * 32) return;
    int k = i & 31;
    int m = i >> 5;
    int u = m / T_DIM, t = m % T_DIM;
    a0[i] = f2b(z[t * 512 + u * 32 + k]);
}

// ---------------------------------------------------------------------------
// Generic bf16 WMMA GEMM: C[M,N] = act(A[M,K] @ B[N,K]^T + bias[N])
// A,B row-major bf16, K multiple of 32, M multiple of 16.
// block = 256 threads (8 waves), wave w -> n-tile blockIdx.y*128 + w*16.
__global__ void gemm_bf16(const u16* __restrict__ A, const u16* __restrict__ B,
                          const float* __restrict__ bias,
                          float* __restrict__ Cf, u16* __restrict__ Ch,
                          int M, int N, int K, int act) {
    int wave = threadIdx.x >> 5;
    int lane = threadIdx.x & 31;
    int m0 = blockIdx.x * 16;
    int n0 = blockIdx.y * 128 + wave * 16;
    if (n0 >= N) return;
    int row = lane & 15;
    int hi  = lane >> 4;

    const u16* Arow = A + (size_t)(m0 + row) * K + hi * 8;
    const u16* Brow = B + (size_t)(n0 + row) * K + hi * 8;

    v8f acc = zero8();
    for (int k0 = 0; k0 < K; k0 += 32) {
        __builtin_prefetch((const void*)(Arow + k0 + 256), 0, 1);
        __builtin_prefetch((const void*)(Brow + k0 + 256), 0, 1);
        Frag a, b;
        a.q[0] = *(const vu4*)(Arow + k0);
        a.q[1] = *(const vu4*)(Arow + k0 + 16);
        b.q[0] = *(const vu4*)(Brow + k0);
        b.q[1] = *(const vu4*)(Brow + k0 + 16);
        acc = __builtin_amdgcn_wmma_f32_16x16x32_bf16(false, a.v, false, b.v,
                                                      (short)0, acc, false, false);
    }
    int n = n0 + row;
    float bv = bias ? bias[n] : 0.0f;
#pragma unroll
    for (int r = 0; r < 8; ++r) {
        int m = m0 + r + 8 * hi;
        float v = acc[r] + bv;
        if (act == 1) v = tanhf_(v);
        if (Cf) Cf[(size_t)m * N + n] = v;
        if (Ch) Ch[(size_t)m * N + n] = f2b(v);
    }
}

// ---------------------------------------------------------------------------
// Cooperative multi-workgroup LSTM scan.
//   grid = H/128 blocks (4 for H=512, 8 for H=1024), block = 256 threads (8 waves).
// Each global wave owns one 16-wide j-tile: it computes the i/f/g/o gate tiles
// for its columns (cell state c stays in accumulator registers), so each WGP
// only streams its 1/N_WG slice of whh from L2 per step. The h vector (bf16,
// 16*H = 16..32KB) is exchanged through L2 each step, double-buffered in
// hglob[t&1], with one device-scope atomic release/acquire barrier per step.
// gx  : [16, T, 4H] f32 (input contribution + bias, precomputed)
// whh : [4H, H] bf16 ; h0 : [16, H] f32 or nullptr
// hseq: [16, T, H] bf16 out ; hT : [16, H] f32 out or nullptr
__global__ void lstm_scan(const float* __restrict__ gx, const u16* __restrict__ whh,
                          const float* __restrict__ h0,
                          u16* __restrict__ hseq, float* __restrict__ hT,
                          u16* __restrict__ hglob, unsigned int* __restrict__ bar,
                          int T, int H) {
    __shared__ __align__(16) u16 h_sh[16 * 1024];   // 32 KB bf16 hidden state

    int tid  = threadIdx.x;
    int wave = tid >> 5;
    int lane = tid & 31;
    int row  = lane & 15;
    int hi   = lane >> 4;
    int H4   = 4 * H;
    unsigned int NWG = gridDim.x;
    int gw = blockIdx.x * 8 + wave;     // global wave id == j-tile id
    int j0 = gw * 16;
    int j  = j0 + row;

    // init hidden state in LDS
    for (int i = tid; i < 16 * H; i += 256)
        h_sh[i] = f2b(h0 ? h0[i] : 0.0f);
    __syncthreads();

    v8f c = zero8();
    const u16* hrow = h_sh + row * H + hi * 8;

    for (int t = 0; t < T; ++t) {
        v8f acc[4];
#pragma unroll
        for (int g = 0; g < 4; ++g) acc[g] = zero8();

        for (int k0 = 0; k0 < H; k0 += 32) {
            Frag a;
            a.q[0] = *(const vu4*)(hrow + k0);
            a.q[1] = *(const vu4*)(hrow + k0 + 16);
#pragma unroll
            for (int g = 0; g < 4; ++g) {
                const u16* Brow = whh + (size_t)(g * H + j0 + row) * H + hi * 8 + k0;
                __builtin_prefetch((const void*)(Brow + 32), 0, 1);
                Frag b;
                b.q[0] = *(const vu4*)(Brow);
                b.q[1] = *(const vu4*)(Brow + 16);
                acc[g] = __builtin_amdgcn_wmma_f32_16x16x32_bf16(
                    false, a.v, false, b.v, (short)0, acc[g], false, false);
            }
        }

        // add precomputed input gates, LSTM cell update (in-register c)
        v8f hv;
        u16* hg = hglob + (size_t)(t & 1) * 16 * H;
#pragma unroll
        for (int r = 0; r < 8; ++r) {
            int m = r + 8 * hi;
            size_t base = ((size_t)m * T + t) * H4;
            float gi = acc[0][r] + gx[base + j0 + row];
            float gf = acc[1][r] + gx[base + H + j0 + row];
            float gg = acc[2][r] + gx[base + 2 * H + j0 + row];
            float go = acc[3][r] + gx[base + 3 * H + j0 + row];
            float cn = sigf(gf) * c[r] + sigf(gi) * tanhf_(gg);
            c[r] = cn;
            hv[r] = sigf(go) * tanhf_(cn);
        }
#pragma unroll
        for (int r = 0; r < 8; ++r) {
            int m = r + 8 * hi;
            u16 hb = f2b(hv[r]);
            hg[m * H + j] = hb;                            // publish own slice
            hseq[((size_t)m * T + t) * H + j] = hb;        // sequence output
            if (hT && t == T - 1) hT[m * H + j] = hv[r];
        }

        // device-scope barrier: release own h slice, acquire everyone's
        __threadfence();
        __syncthreads();
        if (tid == 0) {
            atomicAdd(bar, 1u);
            unsigned int tgt = (unsigned int)(t + 1) * NWG;
            while (*(volatile unsigned int*)bar < tgt)
                __builtin_amdgcn_s_sleep(1);
        }
        __syncthreads();
        __threadfence();

        // reload full h(t) into LDS for step t+1
        {
            const unsigned int* src = (const unsigned int*)hg;
            unsigned int* dst = (unsigned int*)h_sh;
            for (int i = tid; i < 8 * H; i += 256) dst[i] = src[i];
        }
        __syncthreads();
    }
}

// ---------------------------------------------------------------------------
// Softmax over the time axis per (u, o); write transposed to out[b, u*16+o].
// grid = 256 blocks (u*16+o), block = 512 threads (= T).
__global__ void softmax_time(const float* __restrict__ logits, float* __restrict__ out) {
    __shared__ float red[T_DIM];
    int u = blockIdx.x >> 4;
    int o = blockIdx.x & 15;
    int t = threadIdx.x;
    float v = logits[((size_t)u * T_DIM + t) * OUTD + o];
    red[t] = v; __syncthreads();
    for (int s = T_DIM / 2; s > 0; s >>= 1) {
        if (t < s) red[t] = fmaxf(red[t], red[t + s]);
        __syncthreads();
    }
    float mx = red[0]; __syncthreads();
    float e = __expf(v - mx);
    red[t] = e; __syncthreads();
    for (int s = T_DIM / 2; s > 0; s >>= 1) {
        if (t < s) red[t] += red[t + s];
        __syncthreads();
    }
    out[(size_t)t * (U_DIM * OUTD) + u * OUTD + o] = e / red[0];
}

// ---------------------------------------------------------------------------
static inline int cdiv(int a, int b) { return (a + b - 1) / b; }

extern "C" void kernel_launch(void* const* d_in, const int* in_sizes, int n_in,
                              void* d_out, int out_size, void* d_ws, size_t ws_size,
                              hipStream_t stream) {
    (void)in_sizes; (void)n_in; (void)out_size; (void)ws_size;
    const float* z      = (const float*)d_in[0];
    const float* fc0_w  = (const float*)d_in[1];
    const float* fc0_b  = (const float*)d_in[2];
    const float* fc1_w  = (const float*)d_in[3];
    const float* fc1_b  = (const float*)d_in[4];
    const float* c_wih0 = (const float*)d_in[5];
    const float* c_whh0 = (const float*)d_in[6];
    const float* c_b0   = (const float*)d_in[7];
    const float* c_wih1 = (const float*)d_in[8];
    const float* c_whh1 = (const float*)d_in[9];
    const float* c_b1   = (const float*)d_in[10];
    const float* d_wih0 = (const float*)d_in[11];
    const float* d_whh0 = (const float*)d_in[12];
    const float* d_b0   = (const float*)d_in[13];
    const float* d_wih1 = (const float*)d_in[14];
    const float* d_whh1 = (const float*)d_in[15];
    const float* d_b1   = (const float*)d_in[16];
    const float* out_w  = (const float*)d_in[17];
    const float* out_b  = (const float*)d_in[18];
    float* out = (float*)d_out;

    char* base = (char*)d_ws;
    size_t off = 0;
    auto alloc = [&](size_t bytes) -> void* {
        void* p = base + off;
        off += (bytes + 255) & ~(size_t)255;
        return p;
    };

    const int MT = U_DIM * T_DIM;   // 8192

    // bf16 weight mirrors
    u16* wfc0 = (u16*)alloc((size_t)COND * 32 * 2);
    u16* wfc1 = (u16*)alloc((size_t)HID * COND * 2);
    u16* cw0  = (u16*)alloc((size_t)4 * COND * COND * 2);
    u16* ch0  = (u16*)alloc((size_t)4 * COND * COND * 2);
    u16* cw1  = (u16*)alloc((size_t)4 * COND * COND * 2);
    u16* ch1  = (u16*)alloc((size_t)4 * COND * COND * 2);
    u16* dw0  = (u16*)alloc((size_t)4 * HID * COND * 2);
    u16* dh0  = (u16*)alloc((size_t)4 * HID * HID * 2);
    u16* dw1  = (u16*)alloc((size_t)4 * HID * HID * 2);
    u16* dh1  = (u16*)alloc((size_t)4 * HID * HID * 2);
    u16* wout = (u16*)alloc((size_t)OUTD * HID * 2);
    // activations / scratch
    u16*   A0     = (u16*)alloc((size_t)MT * 32 * 2);
    u16*   x0     = (u16*)alloc((size_t)MT * COND * 2);
    float* gx     = (float*)alloc((size_t)MT * 4 * HID * 4);     // reused by all 4 layers
    u16*   h0seq  = (u16*)alloc((size_t)MT * COND * 2);
    u16*   h1seq  = (u16*)alloc((size_t)MT * COND * 2);
    u16*   d0seq  = (u16*)alloc((size_t)MT * HID * 2);
    u16*   d1seq  = (u16*)alloc((size_t)MT * HID * 2);
    float* hT0    = (float*)alloc((size_t)U_DIM * COND * 4);
    float* hT1    = (float*)alloc((size_t)U_DIM * COND * 4);
    u16*   hT0b   = (u16*)alloc((size_t)U_DIM * COND * 2);
    u16*   hT1b   = (u16*)alloc((size_t)U_DIM * COND * 2);
    float* hd0    = (float*)alloc((size_t)U_DIM * HID * 4);
    float* hd1    = (float*)alloc((size_t)U_DIM * HID * 4);
    float* logits = (float*)alloc((size_t)MT * OUTD * 4);
    u16*   hglob  = (u16*)alloc((size_t)2 * 16 * HID * 2);       // double-buffered h exchange
    unsigned int* bars = (unsigned int*)alloc(4 * sizeof(unsigned int));

    auto cvt = [&](const float* s, u16* d, int n) {
        cvt_f32_bf16<<<cdiv(n, 256), 256, 0, stream>>>(s, d, n);
    };
    auto gemm = [&](const u16* A, const u16* B, const float* bias,
                    float* Cf, u16* Ch, int M, int N, int K, int act) {
        dim3 grid(M / 16, cdiv(N, 128));
        gemm_bf16<<<grid, 256, 0, stream>>>(A, B, bias, Cf, Ch, M, N, K, act);
    };
    auto scan = [&](const float* gxp, const u16* whh, const float* h0,
                    u16* hseq, float* hT, unsigned int* bar, int H) {
        lstm_scan<<<H / 128, 256, 0, stream>>>(gxp, whh, h0, hseq, hT, hglob, bar, T_DIM, H);
    };

    // 0) zero the cross-workgroup barrier counters (re-done every launch)
    zero_u32<<<1, 64, 0, stream>>>(bars, 4);

    // 1) weights -> bf16
    cvt(fc0_w,  wfc0, COND * 32);
    cvt(fc1_w,  wfc1, HID * COND);
    cvt(c_wih0, cw0, 4 * COND * COND);
    cvt(c_whh0, ch0, 4 * COND * COND);
    cvt(c_wih1, cw1, 4 * COND * COND);
    cvt(c_whh1, ch1, 4 * COND * COND);
    cvt(d_wih0, dw0, 4 * HID * COND);
    cvt(d_whh0, dh0, 4 * HID * HID);
    cvt(d_wih1, dw1, 4 * HID * HID);
    cvt(d_whh1, dh1, 4 * HID * HID);
    cvt(out_w,  wout, OUTD * HID);

    // 2) pack z -> [U*T, 32] bf16, then fc0: x0 = tanh(A0 @ fc0_w^T + b)
    pack_z<<<cdiv(MT * 32, 256), 256, 0, stream>>>(z, A0);
    gemm(A0, wfc0, fc0_b, nullptr, x0, MT, COND, 32, 1);

    // 3) conductor layer 0
    gemm(x0, cw0, c_b0, gx, nullptr, MT, 4 * COND, COND, 0);
    scan(gx, ch0, nullptr, h0seq, hT0, bars + 0, COND);
    // 4) conductor layer 1
    gemm(h0seq, cw1, c_b1, gx, nullptr, MT, 4 * COND, COND, 0);
    scan(gx, ch1, nullptr, h1seq, hT1, bars + 1, COND);

    // 5) decoder initial hidden: hd = tanh(hT @ fc1_w^T + b)
    cvt(hT0, hT0b, U_DIM * COND);
    cvt(hT1, hT1b, U_DIM * COND);
    gemm(hT0b, wfc1, fc1_b, hd0, nullptr, U_DIM, HID, COND, 1);
    gemm(hT1b, wfc1, fc1_b, hd1, nullptr, U_DIM, HID, COND, 1);

    // 6) decoder layer 0  (input COND -> HID)
    gemm(h1seq, dw0, d_b0, gx, nullptr, MT, 4 * HID, COND, 0);
    scan(gx, dh0, hd0, d0seq, nullptr, bars + 2, HID);
    // 7) decoder layer 1  (HID -> HID)
    gemm(d0seq, dw1, d_b1, gx, nullptr, MT, 4 * HID, HID, 0);
    scan(gx, dh1, hd1, d1seq, nullptr, bars + 3, HID);

    // 8) output projection [8192, 16] and softmax over time axis
    gemm(d1seq, wout, out_b, logits, nullptr, MT, OUTD, HID, 0);
    softmax_time<<<U_DIM * OUTD, T_DIM, 0, stream>>>(logits, out);
}
